// DAGERC_new_2_81990925681280
// MI455X (gfx1250) — matrix operand order repaired and checked by reading, not compiled, not run on hardware
//
#include <hip/hip_runtime.h>
#include <cstdint>
#include <cstddef>

// ---------------------------------------------------------------------------
// DAG-ERC forward for MI455X (gfx1250, wave32, WMMA).
// GEMMs use v_wmma_f32_16x16x32_f16 in the A(row-major) @ W(row-major)^T form
// with 2x2 register blocking (32x32 macro-tile per wave: 4 WMMAs per K-step
// from 2 A-frags + 2 B-frags -> ~16 flop/byte from L2).
// The 8 sequential DAG passes run as persistent cooperative kernels (16 WGs)
// with an atomic grid barrier; x-dependent GRU GEMMs are hoisted out of the
// recurrence into big WMMA GEMMs.
// Scratch use: ~310 MB of d_ws (bump-allocated, deterministic).
// ---------------------------------------------------------------------------

typedef _Float16 half_t;
typedef __attribute__((ext_vector_type(16))) _Float16 v16h;
typedef __attribute__((ext_vector_type(8)))  _Float16 v8h;
typedef __attribute__((ext_vector_type(8)))  float    v8f;

#define BN 64
#define NN 128
#define EE 1024
#define DD 256
#define ROWS (BN * NN)     /* 8192 */
#define BIGF 1.0e30f
#define NWG 16
#define TPB 256

// ---------------------------------------------------------------- device utils
__device__ __forceinline__ float sigmoidf_(float x) { return 1.f / (1.f + expf(-x)); }

__device__ __forceinline__ float wred_sum(float p) {
  p += __shfl_xor(p, 16, 32); p += __shfl_xor(p, 8, 32); p += __shfl_xor(p, 4, 32);
  p += __shfl_xor(p, 2, 32);  p += __shfl_xor(p, 1, 32); return p;
}
__device__ __forceinline__ float wred_max(float p) {
  p = fmaxf(p, __shfl_xor(p, 16, 32)); p = fmaxf(p, __shfl_xor(p, 8, 32));
  p = fmaxf(p, __shfl_xor(p, 4, 32));  p = fmaxf(p, __shfl_xor(p, 2, 32));
  p = fmaxf(p, __shfl_xor(p, 1, 32));  return p;
}

// 32x32 macro-tile: acc[i][j] += A(32xK) @ W(32xK)^T subtiles, fp16 row-major,
// K % 32 == 0.  Fragment layouts (this code's contiguous k order, same for A and B):
//  A-frag: lane l holds row (l&15); halves 0..7 = K[k0+16g..+7], 8..15 =
//          K[k0+16g+8..+7]  (g = l>>4)  -> two contiguous 16B loads.
//  B-frag: lane l holds W row n=(l&15); halves h = K[k0+16g+h] -> one 32B load.
__device__ __forceinline__ void wmma_block2x2(const half_t* __restrict__ A, int lda,
                                              const half_t* __restrict__ W, int ldw,
                                              int K, int lane, v8f acc[2][2]) {
  const int r = lane & 15;
  const int g = lane >> 4;
  const half_t* pa = A + (size_t)r * lda + 16 * g;
  const half_t* pw = W + (size_t)r * ldw + 16 * g;
  const size_t aoff = (size_t)16 * lda;
  const size_t woff = (size_t)16 * ldw;
  for (int k0 = 0; k0 < K; k0 += 32) {
    v16h a0, a1, b0, b1;
    ((v8h*)&a0)[0] = *(const v8h*)(pa + k0);
    ((v8h*)&a0)[1] = *(const v8h*)(pa + k0 + 8);
    ((v8h*)&a1)[0] = *(const v8h*)(pa + aoff + k0);
    ((v8h*)&a1)[1] = *(const v8h*)(pa + aoff + k0 + 8);
    b0 = *(const v16h*)(pw + k0);
    b1 = *(const v16h*)(pw + woff + k0);
    acc[0][0] = __builtin_amdgcn_wmma_f32_16x16x32_f16(false, a0, false, b0, (short)0,
                                                       acc[0][0], false, false);
    acc[0][1] = __builtin_amdgcn_wmma_f32_16x16x32_f16(false, a0, false, b1, (short)0,
                                                       acc[0][1], false, false);
    acc[1][0] = __builtin_amdgcn_wmma_f32_16x16x32_f16(false, a1, false, b0, (short)0,
                                                       acc[1][0], false, false);
    acc[1][1] = __builtin_amdgcn_wmma_f32_16x16x32_f16(false, a1, false, b1, (short)0,
                                                       acc[1][1], false, false);
  }
}

// atomic grid barrier (16 resident WGs); monotonically increasing counter.
__device__ __forceinline__ void grid_sync(volatile unsigned* cnt, unsigned& iter,
                                          unsigned nwg) {
  iter++;
  __syncthreads();
  if (threadIdx.x == 0) {
    __threadfence();
    atomicAdd((unsigned*)cnt, 1u);
    const unsigned target = iter * nwg;
    while (*cnt < target) __builtin_amdgcn_s_sleep(1);
  }
  __syncthreads();
  __threadfence();
}

// ---------------------------------------------------------------- small kernels
__global__ void k_f32_to_f16(const float* __restrict__ s, half_t* __restrict__ d,
                             long long n) {
  long long i = (long long)blockIdx.x * blockDim.x + threadIdx.x;
  if (i < n) d[i] = (half_t)s[i];
}

// at[c*R + r] = a[r*C + c]  (fp32 -> fp16 transpose)
__global__ void k_transpose_f16(const float* __restrict__ a, half_t* __restrict__ at,
                                int R, int C) {
  long long i = (long long)blockIdx.x * blockDim.x + threadIdx.x;
  if (i >= (long long)R * C) return;
  int r = (int)(i / C), c = (int)(i % C);
  at[(size_t)c * R + r] = (half_t)a[i];
}

// pass-frame input: xf/xh[b,i,d] = src[b, rev?N-1-i:i, d]
__global__ void k_pass_prep(const float* __restrict__ src, float* __restrict__ xf,
                            half_t* __restrict__ xh, int rev) {
  long long i = (long long)blockIdx.x * blockDim.x + threadIdx.x;
  if (i >= (long long)BN * NN * DD) return;
  int d = (int)(i & (DD - 1));
  int r = (int)((i >> 8) & (NN - 1));
  int b = (int)(i / (NN * DD));
  int rs = rev ? (NN - 1 - r) : r;
  float v = src[((size_t)b * NN + rs) * DD + d];
  xf[i] = v; xh[i] = (half_t)v;
}

// features (fp32) -> columns [0,1024) of Hfin_h (ld 2048)
__global__ void k_feat_to_hfin(const float* __restrict__ f, half_t* __restrict__ hfin) {
  long long i = (long long)blockIdx.x * blockDim.x + threadIdx.x;
  if (i >= (long long)ROWS * EE) return;
  long long row = i / EE; int c = (int)(i % EE);
  hfin[row * 2048 + c] = (half_t)f[i];
}

// generic (batched) WMMA GEMM: C/Ch[m,n] = act(A(MxK) @ W(NxK)^T + bias[n])
// M, N must be multiples of 32 (true for every call in this model).
__global__ void __launch_bounds__(256)
k_gemm(const half_t* __restrict__ A, int lda, long long sA,
       const half_t* __restrict__ W, int ldw, long long sW,
       const float* __restrict__ bias,
       float* __restrict__ C, int ldc, long long sC,
       half_t* __restrict__ Ch, int ldch, long long sCh,
       int M, int N, int K, int nb, int act) {
  const int lane = threadIdx.x & 31;
  const int gwv = (blockIdx.x * blockDim.x + threadIdx.x) >> 5;
  const int nwaves = (gridDim.x * blockDim.x) >> 5;
  const int tM = M >> 5, tN = N >> 5;   // 32x32 macro-tiles
  const long long tiles = (long long)nb * tM * tN;
  const int r = lane & 15, g = lane >> 4;
  for (long long t = gwv; t < tiles; t += nwaves) {
    const int bat = (int)(t / ((long long)tM * tN));
    const int rem = (int)(t % ((long long)tM * tN));
    const int tm = rem / tN, tn = rem % tN;
    const half_t* Ab = A + (long long)bat * sA + (size_t)tm * 32 * lda;
    const half_t* Wb = W + (long long)bat * sW + (size_t)tn * 32 * ldw;
    v8f acc[2][2] = {{{}, {}}, {{}, {}}};
    wmma_block2x2(Ab, lda, Wb, ldw, K, lane, acc);
#pragma unroll
    for (int im = 0; im < 2; ++im) {
#pragma unroll
      for (int jn = 0; jn < 2; ++jn) {
        const int n = tn * 32 + jn * 16 + r;
        const float bv = bias ? bias[n] : 0.f;
#pragma unroll
        for (int e = 0; e < 8; ++e) {
          const int m = tm * 32 + im * 16 + e + 8 * g;
          float v = acc[im][jn][e] + bv;
          if (act) v = fmaxf(v, 0.f);
          if (C)  C[(long long)bat * sC + (size_t)m * ldc + n] = v;
          if (Ch) Ch[(long long)bat * sCh + (size_t)m * ldch + n] = (half_t)v;
        }
      }
    }
  }
}

// ------------------------------------------------------ persistent DAG pass
__global__ void __launch_bounds__(TPB)
k_dag_pass(const float* __restrict__ Q, const float* __restrict__ GIc,
           const float* __restrict__ GHp, const float* __restrict__ Xf,
           const int* __restrict__ adj, int rev,
           const half_t* __restrict__ cwhh, const half_t* __restrict__ pwih,
           const half_t* __restrict__ wkm,
           const float* __restrict__ cbhh, const float* __restrict__ pbih,
           const float* __restrict__ bk,
           float* __restrict__ H1, float* __restrict__ Kb,
           float* __restrict__ Mb, half_t* __restrict__ Mh,
           float* __restrict__ Gc, float* __restrict__ Gp,
           half_t* __restrict__ hh, unsigned* cnt) {
  const int tid = threadIdx.x;
  const unsigned nwg = gridDim.x;
  const int nthreads = gridDim.x * blockDim.x;
  const int gtid = blockIdx.x * blockDim.x + tid;
  const int lane = tid & 31, wave = tid >> 5;
  const int gwave = gtid >> 5;
  const int nwaves = nthreads >> 5;
  const int r = lane & 15, g = lane >> 4;
  unsigned iter = 0;
  __shared__ float s_w[NN];

  // zero-init H1, Kb (masked softmax tiers read them before being written)
  for (int i = gtid; i < BN * NN * DD; i += nthreads) { H1[i] = 0.f; Kb[i] = 0.f; }
  grid_sync(cnt, iter, nwg);

  for (int i = 0; i < NN; ++i) {
    // ---- Phase A: attention over prefix -> M (4 batches per WG) ----------
    for (int bb = 0; bb < 4; ++bb) {
      const int b = blockIdx.x * 4 + bb;
      const float* qv = Q + ((size_t)b * NN + i) * DD;
      for (int j = wave; j < NN; j += 8) {
        const float* kb = Kb + ((size_t)b * NN + j) * DD;
        float p = 0.f;
        for (int d = lane; d < DD; d += 32) p += kb[d] * qv[d];
        p = wred_sum(p);
        if (lane == 0) {
          const int ii = rev ? (NN - 1 - i) : i;
          const int jj = rev ? (NN - 1 - j) : j;
          const float a = (float)adj[((size_t)b * NN + ii) * NN + jj];
          float lg = p - (1.f - a) * BIGF;
          if (j >= i) lg -= 2.f * BIGF;
          s_w[j] = lg;
        }
      }
      __syncthreads();
      if (wave == 0) {  // softmax over 128 logits (exact fp32, like reference)
        float m = -3.402823466e38f;
        for (int j = lane; j < NN; j += 32) m = fmaxf(m, s_w[j]);
        m = wred_max(m);
        float ssum = 0.f;
        for (int j = lane; j < NN; j += 32) { float e = expf(s_w[j] - m); s_w[j] = e; ssum += e; }
        ssum = wred_sum(ssum);
        const float inv = 1.f / ssum;
        for (int j = lane; j < NN; j += 32) s_w[j] *= inv;
      }
      __syncthreads();
      {
        const int d = tid;  // 256 threads, 256 dims
        const float* h1b = H1 + (size_t)b * NN * DD + d;
        float acc = 0.f;
        for (int j = 0; j < NN; ++j) acc += s_w[j] * h1b[(size_t)j * DD];
        Mb[b * DD + d] = acc;
        Mh[b * DD + d] = (half_t)acc;
      }
      __syncthreads();
    }
    grid_sync(cnt, iter, nwg);

    // ---- Phase B: Gc = Mh@cwhh^T + cbhh ; Gp = Mh@pwih^T + pbih ----------
    // 32x32 macro-tiles: M=64 -> 2 rows; N = [Gc 768 | Gp 768] -> 24 + 24 cols
    for (int t = gwave; t < 2 * 48; t += nwaves) {
      const int tm = t / 48, tn = t % 48;
      const int mat = tn >= 24;
      const int tn2 = tn - (mat ? 24 : 0);
      const half_t* Wp = mat ? pwih : cwhh;
      const float* bp = mat ? pbih : cbhh;
      float* Op = mat ? Gp : Gc;
      v8f acc[2][2] = {{{}, {}}, {{}, {}}};
      wmma_block2x2(Mh + (size_t)tm * 32 * DD, DD, Wp + (size_t)tn2 * 32 * DD, DD, DD,
                    lane, acc);
#pragma unroll
      for (int im = 0; im < 2; ++im)
#pragma unroll
        for (int jn = 0; jn < 2; ++jn) {
          const int n = tn2 * 32 + jn * 16 + r;
          const float bv = bp[n];
#pragma unroll
          for (int e = 0; e < 8; ++e) {
            const int m = tm * 32 + im * 16 + e + 8 * g;
            Op[(size_t)m * 768 + n] = acc[im][jn][e] + bv;
          }
        }
    }
    grid_sync(cnt, iter, nwg);

    // ---- Phase C: GRU-C + GRU-P elementwise, h = C + P --------------------
    for (int idx = gtid; idx < BN * DD; idx += nthreads) {
      const int b = idx >> 8, d = idx & 255;
      const float* gic = GIc + ((size_t)b * NN + i) * 768;
      const float* ghp = GHp + ((size_t)b * NN + i) * 768;
      const float* gc = Gc + (size_t)b * 768;
      const float* gp = Gp + (size_t)b * 768;
      const float m = Mb[idx];
      const float x = Xf[((size_t)b * NN + i) * DD + d];
      // C = gru(x, M): gi from x (precomp), gh from M
      const float r1 = sigmoidf_(gic[d] + gc[d]);
      const float z1 = sigmoidf_(gic[256 + d] + gc[256 + d]);
      const float n1 = tanhf(gic[512 + d] + r1 * gc[512 + d]);
      const float Cv = (1.f - z1) * n1 + z1 * m;
      // P = gru(M, x): gi from M, gh from x (precomp)
      const float r2 = sigmoidf_(gp[d] + ghp[d]);
      const float z2 = sigmoidf_(gp[256 + d] + ghp[256 + d]);
      const float n2 = tanhf(gp[512 + d] + r2 * ghp[512 + d]);
      const float Pv = (1.f - z2) * n2 + z2 * x;
      const float hv = Cv + Pv;
      H1[((size_t)b * NN + i) * DD + d] = hv;
      hh[idx] = (half_t)hv;
    }
    grid_sync(cnt, iter, nwg);

    // ---- Phase D: Kb[:,i] = h @ wk^T + bk (2 x 8 macro-tiles) -------------
    for (int t = gwave; t < 2 * 8; t += nwaves) {
      const int tm = t >> 3, tn = t & 7;
      v8f acc[2][2] = {{{}, {}}, {{}, {}}};
      wmma_block2x2(hh + (size_t)tm * 32 * DD, DD, wkm + (size_t)tn * 32 * DD, DD, DD,
                    lane, acc);
#pragma unroll
      for (int im = 0; im < 2; ++im)
#pragma unroll
        for (int jn = 0; jn < 2; ++jn) {
          const int n = tn * 32 + jn * 16 + r;
          const float bv = bk[n];
#pragma unroll
          for (int e = 0; e < 8; ++e) {
            const int b = tm * 32 + im * 16 + e + 8 * g;
            Kb[((size_t)b * NN + i) * DD + n] = acc[im][jn][e] + bv;
          }
        }
    }
    grid_sync(cnt, iter, nwg);
  }
}

// ----------------------------------------------------------- diff-loss kernels
// column means of concat([f, bwd], -1) viewed as (8192, 512)
__global__ void k_colmean(const float* __restrict__ f, const float* __restrict__ b,
                          float* __restrict__ mean) {
  const int c = blockIdx.x;  // 0..511
  const float* src = (c < DD) ? (f + c) : (b + (c - DD));
  float s = 0.f;
  for (int r = threadIdx.x; r < ROWS; r += 256) s += src[(size_t)r * DD];
  __shared__ float red[256];
  red[threadIdx.x] = s; __syncthreads();
  for (int o = 128; o; o >>= 1) {
    if (threadIdx.x < o) red[threadIdx.x] += red[threadIdx.x + o];
    __syncthreads();
  }
  if (threadIdx.x == 0) mean[c] = red[0] / (float)ROWS;
}

// center columns, L2-normalize rows, write transposed fp16 (512 x 8192)
__global__ void k_normalize_t(const float* __restrict__ f, const float* __restrict__ b,
                              const float* __restrict__ mean, half_t* __restrict__ xt) {
  const int r = blockIdx.x;   // row 0..8191
  const int c = threadIdx.x;  // 0..255, handles c and c+256
  const float v0 = f[(size_t)r * DD + c] - mean[c];
  const float v1 = b[(size_t)r * DD + c] - mean[DD + c];
  __shared__ float red[256];
  red[c] = v0 * v0 + v1 * v1; __syncthreads();
  for (int o = 128; o; o >>= 1) {
    if (c < o) red[c] += red[c + o];
    __syncthreads();
  }
  const float inv = 1.f / (sqrtf(red[0]) + 1e-6f);
  xt[(size_t)c * ROWS + r] = (half_t)(v0 * inv);
  xt[(size_t)(DD + c) * ROWS + r] = (half_t)(v1 * inv);
}

__global__ void k_reduce_sq(const float* __restrict__ g, int n, float scale,
                            float* __restrict__ out) {
  float s = 0.f;
  for (int i = blockIdx.x * blockDim.x + threadIdx.x; i < n; i += gridDim.x * blockDim.x) {
    const float v = g[i]; s += v * v;
  }
  __shared__ float red[256];
  red[threadIdx.x] = s; __syncthreads();
  for (int o = 128; o; o >>= 1) {
    if (threadIdx.x < o) red[threadIdx.x] += red[threadIdx.x + o];
    __syncthreads();
  }
  if (threadIdx.x == 0) atomicAdd(out, red[0] * scale);
}

// concat([f,bwd],-1): outh (B,128,512) fp16 row-major + outt (B,512,128) fp16
__global__ void k_concat_convert(const float* __restrict__ f, const float* __restrict__ b,
                                 half_t* __restrict__ outh, half_t* __restrict__ outt) {
  long long i = (long long)blockIdx.x * blockDim.x + threadIdx.x;
  if (i >= (long long)BN * NN * 512) return;
  const int c = (int)(i % 512);
  const int n = (int)((i / 512) % NN);
  const int bt = (int)(i / (512 * NN));
  const float v = (c < DD) ? f[((size_t)bt * NN + n) * DD + c]
                           : b[((size_t)bt * NN + n) * DD + (c - DD)];
  outh[i] = (half_t)v;
  outt[((size_t)bt * 512 + c) * NN + n] = (half_t)v;
}

// row softmax over 128, fp16 out
__global__ void k_softmax128(const float* __restrict__ s, half_t* __restrict__ a) {
  const int row = blockIdx.x, tid = threadIdx.x;  // 128 threads
  __shared__ float red[128];
  const float x = s[(size_t)row * NN + tid];
  red[tid] = x; __syncthreads();
  for (int o = 64; o; o >>= 1) {
    if (tid < o) red[tid] = fmaxf(red[tid], red[tid + o]);
    __syncthreads();
  }
  const float m = red[0]; __syncthreads();
  const float e = expf(x - m);
  red[tid] = e; __syncthreads();
  for (int o = 64; o; o >>= 1) {
    if (tid < o) red[tid] += red[tid + o];
    __syncthreads();
  }
  a[(size_t)row * NN + tid] = (half_t)(e / red[0]);
}

// logits = h2 @ out_w^T + out_b  (NC=7, scalar dot)
__global__ void k_logits(const float* __restrict__ h2, const float* __restrict__ w,
                         const float* __restrict__ b, float* __restrict__ out) {
  const int t = blockIdx.x * blockDim.x + threadIdx.x;
  if (t >= ROWS * 7) return;
  const int row = t / 7, c = t % 7;
  const float* h = h2 + (size_t)row * DD;
  const float* wc = w + (size_t)c * DD;
  float s = b[c];
  for (int k = 0; k < DD; ++k) s += h[k] * wc[k];
  out[t] = s;
}

// ---------------------------------------------------------------------- host
static inline size_t align_up(size_t x, size_t a) { return (x + a - 1) & ~(a - 1); }

extern "C" void kernel_launch(void* const* d_in, const int* in_sizes, int n_in,
                              void* d_out, int out_size, void* d_ws, size_t ws_size,
                              hipStream_t stream) {
  (void)in_sizes; (void)n_in; (void)out_size; (void)ws_size;
  const float* features = (const float*)d_in[0];
  const int*   adj1     = (const int*)d_in[1];
  const int*   adj2     = (const int*)d_in[2];
  const float* fc1_w    = (const float*)d_in[6];
  const float* fc1_b    = (const float*)d_in[7];
  const float* gw[4][4];  // [gcs,gps,gcl,gpl][wih,whh,bih,bhh]
  for (int t = 0; t < 4; ++t)
    for (int k = 0; k < 4; ++k) gw[t][k] = (const float*)d_in[8 + t * 4 + k];
  const float* gat[2][4];  // [gats,gatl][wq,bq,wk,bk]
  for (int t = 0; t < 2; ++t)
    for (int k = 0; k < 4; ++k) gat[t][k] = (const float*)d_in[24 + t * 4 + k];
  const float* affine1 = (const float*)d_in[32];
  const float* affine2 = (const float*)d_in[33];
  const float* mlp_w1  = (const float*)d_in[34];
  const float* mlp_b1  = (const float*)d_in[35];
  const float* mlp_w2  = (const float*)d_in[36];
  const float* mlp_b2  = (const float*)d_in[37];
  const float* out_w   = (const float*)d_in[38];
  const float* out_b   = (const float*)d_in[39];
  float* out = (float*)d_out;

  // deterministic bump allocator over d_ws
  char* base = (char*)d_ws;
  size_t off = 0;
  auto alloc = [&](size_t bytes) -> void* {
    off = align_up(off, 256);
    void* p = base + off;
    off += bytes;
    return p;
  };
  auto gemm = [&](const half_t* A, int lda, long long sA, const half_t* W, int ldw,
                  long long sW, const float* bias, float* C, int ldc, long long sC,
                  half_t* Ch, int ldch, long long sCh, int M, int N, int K, int nb,
                  int act) {
    k_gemm<<<dim3(1024), dim3(256), 0, stream>>>(A, lda, sA, W, ldw, sW, bias, C, ldc,
                                                 sC, Ch, ldch, sCh, M, N, K, nb, act);
  };
  auto cvt = [&](const float* src, size_t n) -> half_t* {
    half_t* dst = (half_t*)alloc(n * sizeof(half_t));
    k_f32_to_f16<<<dim3((unsigned)((n + 255) / 256)), dim3(256), 0, stream>>>(
        src, dst, (long long)n);
    return dst;
  };

  unsigned* bars = (unsigned*)alloc(256);  // 8 per-pass barrier counters
  hipMemsetAsync(bars, 0, 256, stream);
  hipMemsetAsync(out + ROWS * 7, 0, sizeof(float), stream);  // dloss slot

  // ---- fp16 weight copies --------------------------------------------------
  half_t* fc1_wh = cvt(fc1_w, (size_t)DD * EE);
  half_t *gwih_h[4], *gwhh_h[4];
  for (int t = 0; t < 4; ++t) {
    gwih_h[t] = cvt(gw[t][0], (size_t)2 * 768 * DD);
    gwhh_h[t] = cvt(gw[t][1], (size_t)2 * 768 * DD);
  }
  half_t *wq_h[2], *wk_h[2];
  for (int t = 0; t < 2; ++t) {
    wq_h[t] = cvt(gat[t][0], (size_t)2 * DD * DD);
    wk_h[t] = cvt(gat[t][2], (size_t)2 * DD * DD);
  }
  half_t* aff1t = (half_t*)alloc((size_t)512 * 512 * 2);
  half_t* aff2t = (half_t*)alloc((size_t)512 * 512 * 2);
  k_transpose_f16<<<dim3(1024), dim3(256), 0, stream>>>(affine1, aff1t, 512, 512);
  k_transpose_f16<<<dim3(1024), dim3(256), 0, stream>>>(affine2, aff2t, 512, 512);
  half_t* mlpw1h = cvt(mlp_w1, (size_t)DD * 2048);
  half_t* mlpw2h = cvt(mlp_w2, (size_t)DD * DD);

  // ---- activation scratch --------------------------------------------------
  float*  H0    = (float*)alloc((size_t)ROWS * DD * 4);
  float*  pout[8];
  for (int p = 0; p < 8; ++p) pout[p] = (float*)alloc((size_t)ROWS * DD * 4);
  float*  HinF = (float*)alloc((size_t)ROWS * DD * 4);
  half_t* HinH = (half_t*)alloc((size_t)ROWS * DD * 2);
  float*  Qb   = (float*)alloc((size_t)ROWS * DD * 4);
  float*  GIc  = (float*)alloc((size_t)ROWS * 768 * 4);
  float*  GHp  = (float*)alloc((size_t)ROWS * 768 * 4);
  float*  Kb   = (float*)alloc((size_t)ROWS * DD * 4);
  float*  Mb   = (float*)alloc((size_t)BN * DD * 4);
  half_t* Mh   = (half_t*)alloc((size_t)BN * DD * 2);
  float*  Gc   = (float*)alloc((size_t)BN * 768 * 4);
  float*  Gp   = (float*)alloc((size_t)BN * 768 * 4);
  half_t* hh   = (half_t*)alloc((size_t)BN * DD * 2);
  half_t* Hfin = (half_t*)alloc((size_t)ROWS * 2048 * 2);
  float*  meanX = (float*)alloc(512 * 4);
  float*  meanY = (float*)alloc(512 * 4);
  half_t* Xt   = (half_t*)alloc((size_t)512 * ROWS * 2);
  half_t* Yt   = (half_t*)alloc((size_t)512 * ROWS * 2);
  float*  Gram = (float*)alloc((size_t)512 * 512 * 4);
  half_t* HSh  = (half_t*)alloc((size_t)ROWS * 512 * 2);
  half_t* HSt  = (half_t*)alloc((size_t)ROWS * 512 * 2);
  half_t* HLh  = (half_t*)alloc((size_t)ROWS * 512 * 2);
  half_t* HLt  = (half_t*)alloc((size_t)ROWS * 512 * 2);
  half_t* T1h  = (half_t*)alloc((size_t)ROWS * 512 * 2);
  half_t* T2h  = (half_t*)alloc((size_t)ROWS * 512 * 2);
  float*  Sb   = (float*)alloc((size_t)BN * NN * NN * 4);
  half_t* A1h  = (half_t*)alloc((size_t)BN * NN * NN * 2);
  half_t* A2h  = (half_t*)alloc((size_t)BN * NN * NN * 2);
  half_t* h1h  = (half_t*)alloc((size_t)ROWS * DD * 2);
  float*  h2f  = (float*)alloc((size_t)ROWS * DD * 4);

  // ---- fc1: H0 = relu(features @ fc1_w^T + b) -----------------------------
  // features fp16 live in Hfin columns [0,1024) (also the MLP input later)
  k_feat_to_hfin<<<dim3((unsigned)((ROWS * (long long)EE + 255) / 256)), dim3(256), 0,
                   stream>>>(features, Hfin);
  gemm(Hfin, 2048, 0, fc1_wh, EE, 0, fc1_b, H0, DD, 0, nullptr, 0, 0,
       ROWS, DD, EE, 1, /*relu*/ 1);

  // ---- 8 DAG passes -------------------------------------------------------
  const int chanOf[8]  = {0, 0, 0, 0, 1, 1, 1, 1};
  const int layerOf[8] = {0, 0, 1, 1, 0, 0, 1, 1};
  const int revOf[8]   = {0, 1, 0, 0, 0, 1, 0, 0};
  const float* srcOf[8] = {H0, H0, pout[0], pout[1], H0, H0, pout[4], pout[5]};
  for (int p = 0; p < 8; ++p) {
    const int ch = chanOf[p], l = layerOf[p];
    const int gc = (ch == 0) ? 0 : 2;  // C-GRU params (gcs / gcl)
    const int gp = (ch == 0) ? 1 : 3;  // P-GRU params (gps / gpl)
    const int* adjc = (ch == 0) ? adj1 : adj2;
    k_pass_prep<<<dim3((unsigned)((BN * NN * DD + 255) / 256)), dim3(256), 0, stream>>>(
        srcOf[p], HinF, HinH, revOf[p]);
    // hoisted x-dependent GEMMs: q, gi(C), gh(P)
    gemm(HinH, DD, 0, wq_h[ch] + (size_t)l * DD * DD, DD, 0, gat[ch][1] + l * DD,
         Qb, DD, 0, nullptr, 0, 0, ROWS, DD, DD, 1, 0);
    gemm(HinH, DD, 0, gwih_h[gc] + (size_t)l * 768 * DD, DD, 0, gw[gc][2] + l * 768,
         GIc, 768, 0, nullptr, 0, 0, ROWS, 768, DD, 1, 0);
    gemm(HinH, DD, 0, gwhh_h[gp] + (size_t)l * 768 * DD, DD, 0, gw[gp][3] + l * 768,
         GHp, 768, 0, nullptr, 0, 0, ROWS, 768, DD, 1, 0);
    k_dag_pass<<<dim3(NWG), dim3(TPB), 0, stream>>>(
        Qb, GIc, GHp, HinF, adjc, revOf[p],
        gwhh_h[gc] + (size_t)l * 768 * DD, gwih_h[gp] + (size_t)l * 768 * DD,
        wk_h[ch] + (size_t)l * DD * DD,
        gw[gc][3] + l * 768, gw[gp][2] + l * 768, gat[ch][3] + l * DD,
        pout[p], Kb, Mb, Mh, Gc, Gp, hh, bars + p);
  }

  // ---- diff loss: BETA * sum_l mean((Xs_l^T Yl_l)^2) ----------------------
  for (int l = 0; l < 2; ++l) {
    const float* fA = pout[l ? 2 : 0];  const float* bA = pout[l ? 3 : 1];  // cs[l]
    const float* fB = pout[l ? 6 : 4];  const float* bB = pout[l ? 7 : 5];  // cl[l]
    k_colmean<<<dim3(512), dim3(256), 0, stream>>>(fA, bA, meanX);
    k_colmean<<<dim3(512), dim3(256), 0, stream>>>(fB, bB, meanY);
    k_normalize_t<<<dim3(ROWS), dim3(256), 0, stream>>>(fA, bA, meanX, Xt);
    k_normalize_t<<<dim3(ROWS), dim3(256), 0, stream>>>(fB, bB, meanY, Yt);
    gemm(Xt, ROWS, 0, Yt, ROWS, 0, nullptr, Gram, 512, 0, nullptr, 0, 0,
         512, 512, ROWS, 1, 0);
    k_reduce_sq<<<dim3(256), dim3(256), 0, stream>>>(Gram, 512 * 512,
                                                     0.3f / (512.f * 512.f),
                                                     out + ROWS * 7);
  }

  // ---- cross attention ----------------------------------------------------
  const unsigned ccb = (unsigned)(((long long)BN * NN * 512 + 255) / 256);
  k_concat_convert<<<dim3(ccb), dim3(256), 0, stream>>>(pout[2], pout[3], HSh, HSt);
  k_concat_convert<<<dim3(ccb), dim3(256), 0, stream>>>(pout[6], pout[7], HLh, HLt);
  gemm(HSh, 512, 0, aff1t, 512, 0, nullptr, nullptr, 0, 0, T1h, 512, 0,
       ROWS, 512, 512, 1, 0);
  gemm(HLh, 512, 0, aff2t, 512, 0, nullptr, nullptr, 0, 0, T2h, 512, 0,
       ROWS, 512, 512, 1, 0);
  // S1[b] = T1[b] @ HL[b]^T ; A1 = softmax ; HSn = A1 @ HL -> Hfin[:,1024:1536]
  gemm(T1h, 512, NN * 512, HLh, 512, NN * 512, nullptr, Sb, NN, NN * NN,
       nullptr, 0, 0, NN, NN, 512, BN, 0);
  k_softmax128<<<dim3(ROWS), dim3(128), 0, stream>>>(Sb, A1h);
  gemm(A1h, NN, NN * NN, HLt, NN, 512 * NN, nullptr, nullptr, 0, 0,
       Hfin + 1024, 2048, (long long)NN * 2048, NN, 512, NN, BN, 0);
  // S2[b] = T2[b] @ HS[b]^T ; A2 = softmax ; HLn = A2 @ HS -> Hfin[:,1536:2048]
  gemm(T2h, 512, NN * 512, HSh, 512, NN * 512, nullptr, Sb, NN, NN * NN,
       nullptr, 0, 0, NN, NN, 512, BN, 0);
  k_softmax128<<<dim3(ROWS), dim3(128), 0, stream>>>(Sb, A2h);
  gemm(A2h, NN, NN * NN, HSt, NN, 512 * NN, nullptr, nullptr, 0, 0,
       Hfin + 1536, 2048, (long long)NN * 2048, NN, 512, NN, BN, 0);

  // ---- MLP head -----------------------------------------------------------
  gemm(Hfin, 2048, 0, mlpw1h, 2048, 0, mlp_b1, nullptr, 0, 0, h1h, DD, 0,
       ROWS, DD, 2048, 1, 1);
  gemm(h1h, DD, 0, mlpw2h, DD, 0, mlp_b2, h2f, DD, 0, nullptr, 0, 0,
       ROWS, DD, DD, 1, 1);
  k_logits<<<dim3((unsigned)((ROWS * 7 + 255) / 256)), dim3(256), 0, stream>>>(
      h2f, out_w, out_b, out);
}
